// CausalNet_59115929862442
// MI455X (gfx1250) — matrix operand order repaired and checked by reading, not compiled, not run on hardware
//
#include <hip/hip_runtime.h>
#include <hip/hip_bf16.h>
#include <stdint.h>

// ---------------------------------------------------------------------------
// CausalNet forward for MI455X (gfx1250): bf16 WMMA + TDM (tensor_load_to_lds)
// Shapes: B=64, N=4, P=128, d=h=256, totP=512, feat=131072, C=2.
// ---------------------------------------------------------------------------

#define BB   64
#define NN   4
#define PP   128
#define DD   256
#define HH   256
#define TOTP 512
#define FEAT 131072

typedef __bf16 bf16_t;
typedef __bf16 v16bf __attribute__((ext_vector_type(16)));
typedef float  v8f   __attribute__((ext_vector_type(8)));

#define WMMA_BF16(a, b, c) \
  __builtin_amdgcn_wmma_f32_16x16x32_bf16(false, (a), false, (b), (short)0, (c), false, false)

#if __has_builtin(__builtin_amdgcn_tensor_load_to_lds)
#define HAS_TDM 1
#else
#define HAS_TDM 0
#endif

typedef unsigned int tdm_v4u __attribute__((ext_vector_type(4)));
typedef int          tdm_v4i __attribute__((ext_vector_type(4)));
typedef int          tdm_v8i __attribute__((ext_vector_type(8)));

// Issue one 2D TDM tile load (global -> LDS).  D# per CDNA5 ISA ch.8:
//  group0: [1:0]count=1 | [63:32]lds_addr | [120:64]global_addr | [127:126]type=2
//  group1: [17:16]data_size | [79:48]tensor_dim0 | [111:80]tensor_dim1
//          | [127:112]tile_dim0 | [143:128]tile_dim1 | [207:160]dim0_stride
// Caller: one wave only; TENSORcnt is per-wave.
static __device__ __forceinline__ void tdm_load_2d(uint32_t lds_off, const void* gptr,
                                                   uint32_t td0, uint32_t td1,
                                                   uint32_t tile0, uint32_t tile1,
                                                   uint32_t stride0, uint32_t dsz_code) {
#if HAS_TDM
  uint64_t ga = (uint64_t)(uintptr_t)gptr;
  tdm_v4u g0;
  g0.x = 1u;                                    // count=1 (valid user descriptor)
  g0.y = lds_off;                               // lds_addr (bytes)
  g0.z = (uint32_t)ga;                          // global_addr[31:0]
  g0.w = ((uint32_t)(ga >> 32) & 0x01FFFFFFu)   // global_addr[56:32]
         | (2u << 30);                          // type = 2 ("image")
  tdm_v8i g1;
  g1[0] = (int)(dsz_code << 16);                           // data_size
  g1[1] = (int)((td0 & 0xFFFFu) << 16);                    // tensor_dim0 lo
  g1[2] = (int)((td0 >> 16) | ((td1 & 0xFFFFu) << 16));    // dim0 hi | dim1 lo
  g1[3] = (int)((td1 >> 16) | ((tile0 & 0xFFFFu) << 16));  // dim1 hi | tile_dim0
  g1[4] = (int)(tile1 & 0xFFFFu);                          // tile_dim1 (tile_dim2=0)
  g1[5] = (int)stride0;                                    // tensor_dim0_stride lo
  g1[6] = 0;
  g1[7] = 0;
  tdm_v4i z4 = {0, 0, 0, 0};
#if defined(__clang_major__) && (__clang_major__ >= 23)
  tdm_v8i z8 = {0, 0, 0, 0, 0, 0, 0, 0};
  __builtin_amdgcn_tensor_load_to_lds(g0, g1, z4, z4, z8, 0);
#else
  __builtin_amdgcn_tensor_load_to_lds(g0, g1, z4, z4, 0);
#endif
#endif
}

static __device__ __forceinline__ void tdm_wait() {
#if HAS_TDM
  __builtin_amdgcn_s_wait_tensorcnt(0);
#endif
}

// Cooperative fallback staging (only compiled in when TDM builtin is absent).
template <typename T>
static __device__ __forceinline__ void stage_fallback(T* dst, const T* src,
                                                      int tile0, int tile1, int stride,
                                                      int tid) {
#if !HAS_TDM
  for (int i = tid; i < tile0 * tile1; i += 256) {
    int r = i / tile0, c = i - r * tile0;
    dst[i] = src[(size_t)r * stride + c];
  }
#endif
}

static __device__ __forceinline__ float to_f(float x)  { return x; }
static __device__ __forceinline__ float to_f(bf16_t x) { return (float)x; }

static __device__ __forceinline__ v8f vzero8() {
  v8f z;
#pragma unroll
  for (int i = 0; i < 8; ++i) z[i] = 0.0f;
  return z;
}

// A fragment (16x32, bf16), A row-major, leading dim lda.
template <typename T>
static __device__ __forceinline__ v16bf load_a_frag(const T* __restrict__ A, int lda, int lane) {
  const int m = lane & 15, half = lane >> 4;
  const T* r = A + (size_t)m * lda + half * 16;
  v16bf a;
#pragma unroll
  for (int j = 0; j < 8; ++j) {
    a[j]     = (bf16_t)to_f(r[j]);
    a[j + 8] = (bf16_t)to_f(r[8 + j]);
  }
  return a;
}

// B fragment (32x16), B row-major [K x N], leading dim ldb.
template <typename T>
static __device__ __forceinline__ v16bf load_b_frag(const T* __restrict__ Bm, int ldb, int lane) {
  const int n = lane & 15, half = lane >> 4;
  const T* c = Bm + (size_t)(half * 16) * ldb + n;
  v16bf b;
#pragma unroll
  for (int j = 0; j < 16; ++j) b[j] = (bf16_t)to_f(c[(size_t)j * ldb]);
  return b;
}

// B fragment from B^T row-major (row n contiguous) -- used for X·Xᵀ.
static __device__ __forceinline__ v16bf load_bt_frag(const bf16_t* __restrict__ Bt, int ldr, int lane) {
  const int n = lane & 15, half = lane >> 4;
  const bf16_t* r = Bt + (size_t)n * ldr + half * 16;
  v16bf b;
#pragma unroll
  for (int j = 0; j < 16; ++j) b[j] = r[j];
  return b;
}

// ---------------------------------------------------------------------------
__global__ __launch_bounds__(256) void k_zero(float* __restrict__ p, int n) {
  for (int i = blockIdx.x * blockDim.x + threadIdx.x; i < n; i += gridDim.x * blockDim.x)
    p[i] = 0.0f;
}

// Row-wise L2 normalize (F.normalize, eps=1e-12): x[32768,256] f32 -> xn bf16
__global__ __launch_bounds__(256) void k_norm(const float* __restrict__ x, bf16_t* __restrict__ xn) {
  __shared__ float red[256];
  const int row = blockIdx.x, tid = threadIdx.x;
  float v = x[(size_t)row * DD + tid];
  red[tid] = v * v;
  __syncthreads();
  for (int s = 128; s > 0; s >>= 1) {
    if (tid < s) red[tid] += red[tid + s];
    __syncthreads();
  }
  float nrm = sqrtf(red[0]);
  float sc = 1.0f / fmaxf(nrm, 1e-12f);
  xn[(size_t)row * DD + tid] = (bf16_t)(v * sc);
}

// Per (b,n): TDM-stage Xn block (128x256 bf16, 64 KiB) into LDS, S = Xn·Xnᵀ
// via WMMA, fuse mask/diag, degree rowsums via shfl_xor, write An bf16.
__global__ __launch_bounds__(256) void k_sim_an(const bf16_t* __restrict__ xn,
                                                const float* __restrict__ mask,
                                                bf16_t* __restrict__ An) {
  __shared__ bf16_t Xs[PP * DD];  // 64 KiB
  __shared__ float sdeg[PP];
  __shared__ float sdinv[PP];
  const int bn = blockIdx.x;
  const int tid = threadIdx.x, lane = tid & 31, wave = tid >> 5;
  const int lid = lane & 15, half = lane >> 4;
  const bf16_t* X = xn + (size_t)bn * PP * DD;

  if (tid < 32) {
    tdm_load_2d((uint32_t)(size_t)(void*)Xs, X, DD, PP, DD, PP, DD, /*2B*/ 1);
    tdm_wait();
  }
  stage_fallback(Xs, X, DD, PP, DD, tid);
  __syncthreads();

  v8f acc[8];
#pragma unroll
  for (int t = 0; t < 8; ++t) acc[t] = vzero8();

  for (int k0 = 0; k0 < DD; k0 += 32) {
    v16bf a = load_a_frag(Xs + (size_t)(wave * 16) * DD + k0, DD, lane);
#pragma unroll
    for (int t = 0; t < 8; ++t) {
      v16bf b = load_bt_frag(Xs + (size_t)(t * 16) * DD + k0, DD, lane);
      acc[t] = WMMA_BF16(a, b, acc[t]);
    }
  }

  float rs[8];
#pragma unroll
  for (int i = 0; i < 8; ++i) rs[i] = 0.0f;
#pragma unroll
  for (int t = 0; t < 8; ++t) {
#pragma unroll
    for (int i = 0; i < 8; ++i) {
      int m = wave * 16 + i + 8 * half;
      int n = t * 16 + lid;
      float v = (acc[t][i] + 1.0f) * 0.5f * mask[m * PP + n];
      if (m == n) v = 1.0f;  // edges diag zeroed, then +I
      acc[t][i] = v;
      rs[i] += v;
    }
  }
#pragma unroll
  for (int i = 0; i < 8; ++i) {
    float r = rs[i];
    r += __shfl_xor(r, 1);
    r += __shfl_xor(r, 2);
    r += __shfl_xor(r, 4);
    r += __shfl_xor(r, 8);
    if (lid == i) sdeg[wave * 16 + i + 8 * half] = r;
  }
  __syncthreads();
  if (tid < PP) {
    float dg = sdeg[tid];
    sdinv[tid] = dg > 0.0f ? rsqrtf(fmaxf(dg, 1e-12f)) : 0.0f;
  }
  __syncthreads();

  bf16_t* Ab = An + (size_t)bn * PP * PP;
#pragma unroll
  for (int t = 0; t < 8; ++t) {
#pragma unroll
    for (int i = 0; i < 8; ++i) {
      int m = wave * 16 + i + 8 * half;
      int n = t * 16 + lid;
      Ab[(size_t)m * PP + n] = (bf16_t)(sdinv[m] * acc[t][i] * sdinv[n]);
    }
  }
}

// out[32768,256](bf16) = A[32768,256] * W[256,256](f32) + bias.
// W column-panel (256x64 f32, 64 KiB) staged once per block via TDM.
template <typename AT>
__global__ __launch_bounds__(256) void k_gemm_xw(const AT* __restrict__ A,
                                                 const float* __restrict__ W,
                                                 const float* __restrict__ bias,
                                                 bf16_t* __restrict__ out) {
  __shared__ float Ws[DD * 64];  // 64 KiB
  const int tid = threadIdx.x, lane = tid & 31, wave = tid >> 5;
  const int lid = lane & 15, half = lane >> 4;
  const int rowbase = blockIdx.x * 128 + wave * 16;
  const int colbase = blockIdx.y * 64;

  if (tid < 32) {
    tdm_load_2d((uint32_t)(size_t)(void*)Ws, W + colbase, HH, DD, 64, DD, HH, /*4B*/ 2);
    tdm_wait();
  }
  stage_fallback(Ws, W + colbase, 64, DD, HH, tid);
  __syncthreads();

  v8f acc[4];
#pragma unroll
  for (int t = 0; t < 4; ++t) acc[t] = vzero8();

  for (int k0 = 0; k0 < DD; k0 += 32) {
    v16bf a = load_a_frag(A + (size_t)rowbase * DD + k0, DD, lane);
#pragma unroll
    for (int t = 0; t < 4; ++t) {
      v16bf b = load_b_frag(Ws + (size_t)k0 * 64 + t * 16, 64, lane);
      acc[t] = WMMA_BF16(a, b, acc[t]);
    }
  }
#pragma unroll
  for (int t = 0; t < 4; ++t) {
#pragma unroll
    for (int i = 0; i < 8; ++i) {
      int m = rowbase + i + 8 * half;
      int n = colbase + t * 16 + lid;
      out[(size_t)m * HH + n] = (bf16_t)(acc[t][i] + bias[n]);
    }
  }
}

// pre = blockdiag(An) @ xw per (b,n), split into 2 column halves.
// xw column-panel (128x128 bf16, 32 KiB) staged via TDM; BN sums fused.
__global__ __launch_bounds__(256) void k_gemm_an(const bf16_t* __restrict__ An,
                                                 const bf16_t* __restrict__ xw,
                                                 bf16_t* __restrict__ pre,
                                                 float* __restrict__ gsum,
                                                 float* __restrict__ gsumsq) {
  __shared__ bf16_t Bs[PP * 128];  // 32 KiB
  __shared__ float ssum[128];
  __shared__ float ssq[128];
  const int bn = blockIdx.x;
  const int hsplit = blockIdx.y;
  const int tid = threadIdx.x, lane = tid & 31, wave = tid >> 5;
  const int lid = lane & 15, half = lane >> 4;
  const bf16_t* Ab = An + (size_t)bn * PP * PP;
  const bf16_t* Xb = xw + (size_t)bn * PP * HH;
  const int colbase = hsplit * 128;

  if (tid < 128) ssum[tid] = 0.0f;
  else ssq[tid - 128] = 0.0f;
  if (tid < 32) {
    tdm_load_2d((uint32_t)(size_t)(void*)Bs, Xb + colbase, HH, PP, 128, PP, HH, /*2B*/ 1);
    tdm_wait();
  }
  stage_fallback(Bs, Xb + colbase, 128, PP, HH, tid);
  __syncthreads();

  v8f acc[8];
#pragma unroll
  for (int t = 0; t < 8; ++t) acc[t] = vzero8();

  for (int k0 = 0; k0 < PP; k0 += 32) {
    v16bf a = load_a_frag(Ab + (size_t)(wave * 16) * PP + k0, PP, lane);
#pragma unroll
    for (int t = 0; t < 8; ++t) {
      v16bf b = load_b_frag(Bs + (size_t)k0 * 128 + t * 16, 128, lane);
      acc[t] = WMMA_BF16(a, b, acc[t]);
    }
  }

  bf16_t* Ob = pre + (size_t)bn * PP * HH;
#pragma unroll
  for (int t = 0; t < 8; ++t) {
    float s = 0.0f, q = 0.0f;
#pragma unroll
    for (int i = 0; i < 8; ++i) {
      int m = wave * 16 + i + 8 * half;
      int n = colbase + t * 16 + lid;
      float v = acc[t][i];
      Ob[(size_t)m * HH + n] = (bf16_t)v;
      s += v;
      q += v * v;
    }
    s += __shfl_xor(s, 16);
    q += __shfl_xor(q, 16);
    if (lane < 16) {
      atomicAdd(&ssum[t * 16 + lid], s);
      atomicAdd(&ssq[t * 16 + lid], q);
    }
  }
  __syncthreads();
  if (tid < 128) atomicAdd(&gsum[colbase + tid], ssum[tid]);
  else atomicAdd(&gsumsq[colbase + tid - 128], ssq[tid - 128]);
}

// Fold BN statistics into per-channel scale/shift.
__global__ __launch_bounds__(256) void k_bnfin(const float* __restrict__ gsum,
                                               const float* __restrict__ gsumsq,
                                               const float* __restrict__ g,
                                               const float* __restrict__ be,
                                               float* __restrict__ scale,
                                               float* __restrict__ shift) {
  int c = threadIdx.x;
  if (c < HH) {
    const float cnt = (float)(BB * TOTP);
    float m = gsum[c] / cnt;
    float v = gsumsq[c] / cnt - m * m;
    float r = rsqrtf(v + 1e-5f);
    float sc = g[c] * r;
    scale[c] = sc;
    shift[c] = be[c] - m * sc;
  }
}

// act = relu(scale*pre + shift), elementwise, channel = idx & 255
__global__ __launch_bounds__(256) void k_bnapply(const bf16_t* __restrict__ pre,
                                                 const float* __restrict__ scale,
                                                 const float* __restrict__ shift,
                                                 bf16_t* __restrict__ act, int n) {
  for (int i = blockIdx.x * blockDim.x + threadIdx.x; i < n; i += gridDim.x * blockDim.x) {
    int c = i & (HH - 1);
    float y = scale[c] * (float)pre[i] + shift[c];
    act[i] = (bf16_t)fmaxf(y, 0.0f);
  }
}

// z1pre[64,128] += graph[64,131072] @ Wm1[131072,128], split-K over 64 WGs.
// Wm1 streamed through LDS in 128x128 f32 chunks via TDM.
__global__ __launch_bounds__(256) void k_head(const bf16_t* __restrict__ graph,
                                              const float* __restrict__ Wm1,
                                              float* __restrict__ z1pre) {
  __shared__ float Bs[128 * 128];  // 64 KiB
  const int tid = threadIdx.x, lane = tid & 31, wave = tid >> 5;
  const int lid = lane & 15, half = lane >> 4;
  const int mstrip = (wave & 3) * 16;
  const int nbase = (wave >> 2) * 64;
  const int kbase = blockIdx.x * 2048;

  v8f acc[4];
#pragma unroll
  for (int t = 0; t < 4; ++t) acc[t] = vzero8();

  for (int kc = 0; kc < 2048; kc += 128) {
    if (tid < 32) {
      tdm_load_2d((uint32_t)(size_t)(void*)Bs, Wm1 + (size_t)(kbase + kc) * 128,
                  128, 128, 128, 128, 128, /*4B*/ 2);
      tdm_wait();
    }
    stage_fallback(Bs, Wm1 + (size_t)(kbase + kc) * 128, 128, 128, 128, tid);
    __syncthreads();

#pragma unroll
    for (int kr = 0; kr < 128; kr += 32) {
      const int k0 = kbase + kc + kr;
      __builtin_prefetch(graph + (size_t)mstrip * FEAT + k0 + 128, 0, 1);
      v16bf a = load_a_frag(graph + (size_t)mstrip * FEAT + k0, FEAT, lane);
#pragma unroll
      for (int t = 0; t < 4; ++t) {
        v16bf b = load_b_frag(Bs + (size_t)kr * 128 + nbase + t * 16, 128, lane);
        acc[t] = WMMA_BF16(a, b, acc[t]);
      }
    }
    __syncthreads();
  }
#pragma unroll
  for (int t = 0; t < 4; ++t) {
#pragma unroll
    for (int i = 0; i < 8; ++i) {
      int m = mstrip + i + 8 * half;
      int n = nbase + t * 16 + lid;
      atomicAdd(&z1pre[m * 128 + n], acc[t][i]);
    }
  }
}

// BN1d(+bm1) -> relu -> Wm2+bm2 -> BN1d -> relu -> Wm3+bm3, single workgroup
__global__ __launch_bounds__(256) void k_tail(const float* __restrict__ z1pre,
                                              const float* bm1, const float* gm1, const float* bem1,
                                              const float* Wm2, const float* bm2,
                                              const float* gm2, const float* bem2,
                                              const float* Wm3, const float* bm3,
                                              float* __restrict__ out) {
  __shared__ float z1[BB * 128];
  __shared__ float z2[BB * 64];
  __shared__ float sc1[128], sh1[128], sc2[64], sh2[64];
  const int tid = threadIdx.x;

  if (tid < 128) {
    float s = 0.0f, q = 0.0f;
    for (int r = 0; r < BB; ++r) {
      float v = z1pre[r * 128 + tid] + bm1[tid];
      s += v;
      q += v * v;
    }
    float m = s / BB, var = q / BB - m * m, rs = rsqrtf(var + 1e-5f);
    float sc = gm1[tid] * rs;
    sc1[tid] = sc;
    sh1[tid] = bem1[tid] - m * sc;
  }
  __syncthreads();
  for (int i = tid; i < BB * 128; i += 256) {
    int f = i & 127;
    float v = z1pre[i] + bm1[f];
    z1[i] = fmaxf(sc1[f] * v + sh1[f], 0.0f);
  }
  __syncthreads();
  for (int i = tid; i < BB * 64; i += 256) {
    int r = i >> 6, c = i & 63;
    float s = bm2[c];
    for (int k = 0; k < 128; ++k) s += z1[r * 128 + k] * Wm2[k * 64 + c];
    z2[i] = s;
  }
  __syncthreads();
  if (tid < 64) {
    float s = 0.0f, q = 0.0f;
    for (int r = 0; r < BB; ++r) {
      float v = z2[r * 64 + tid];
      s += v;
      q += v * v;
    }
    float m = s / BB, var = q / BB - m * m, rs = rsqrtf(var + 1e-5f);
    float sc = gm2[tid] * rs;
    sc2[tid] = sc;
    sh2[tid] = bem2[tid] - m * sc;
  }
  __syncthreads();
  for (int i = tid; i < BB * 64; i += 256) {
    int c = i & 63;
    z2[i] = fmaxf(sc2[c] * z2[i] + sh2[c], 0.0f);
  }
  __syncthreads();
  if (tid < BB * 2) {
    int r = tid >> 1, c = tid & 1;
    float s = bm3[c];
    for (int k = 0; k < 64; ++k) s += z2[r * 64 + k] * Wm3[k * 2 + c];
    out[tid] = s;
  }
}

// ---------------------------------------------------------------------------
extern "C" void kernel_launch(void* const* d_in, const int* in_sizes, int n_in,
                              void* d_out, int out_size, void* d_ws, size_t ws_size,
                              hipStream_t stream) {
  const float* x    = (const float*)d_in[0];
  const float* mask = (const float*)d_in[1];
  const float* W1   = (const float*)d_in[2];
  const float* b1   = (const float*)d_in[3];
  const float* g1   = (const float*)d_in[4];
  const float* be1  = (const float*)d_in[5];
  const float* W2   = (const float*)d_in[6];
  const float* b2   = (const float*)d_in[7];
  const float* g2   = (const float*)d_in[8];
  const float* be2  = (const float*)d_in[9];
  const float* Wm1  = (const float*)d_in[10];
  const float* bm1  = (const float*)d_in[11];
  const float* gm1  = (const float*)d_in[12];
  const float* bem1 = (const float*)d_in[13];
  const float* Wm2  = (const float*)d_in[14];
  const float* bm2  = (const float*)d_in[15];
  const float* gm2  = (const float*)d_in[16];
  const float* bem2 = (const float*)d_in[17];
  const float* Wm3  = (const float*)d_in[18];
  const float* bm3  = (const float*)d_in[19];
  float* out = (float*)d_out;

  char* ws = (char*)d_ws;
  bf16_t* xn  = (bf16_t*)(ws + 0);          // 16 MiB; reused as xw after k_sim_an
  bf16_t* An  = (bf16_t*)(ws + 16777216);   // 8 MiB
  bf16_t* pre = (bf16_t*)(ws + 25165824);   // 16 MiB
  bf16_t* act = (bf16_t*)(ws + 41943040);   // 16 MiB
  float* stats  = (float*)(ws + 58720256);  // 4 KiB: sum/sumsq/scale/shift
  float* gsum   = stats;
  float* gsumsq = stats + 256;
  float* scale  = stats + 512;
  float* shift  = stats + 768;
  float* z1pre  = (float*)(ws + 58724352);  // 32 KiB
  bf16_t* xw = xn;  // alias: xn consumed by k_sim_an before first k_gemm_xw

  const int nelem = BB * TOTP * HH;

  // stage 1: dynamic normalized adjacency
  k_norm<<<dim3(BB * TOTP), 256, 0, stream>>>(x, xn);
  k_sim_an<<<dim3(BB * NN), 256, 0, stream>>>(xn, mask, An);

  // GCN layer 1
  k_zero<<<dim3(2), 256, 0, stream>>>(gsum, 512);
  k_gemm_xw<float><<<dim3(256, 4), 256, 0, stream>>>(x, W1, b1, xw);
  k_gemm_an<<<dim3(BB * NN, 2), 256, 0, stream>>>(An, xw, pre, gsum, gsumsq);
  k_bnfin<<<dim3(1), 256, 0, stream>>>(gsum, gsumsq, g1, be1, scale, shift);
  k_bnapply<<<dim3(2048), 256, 0, stream>>>(pre, scale, shift, act, nelem);

  // GCN layer 2
  k_zero<<<dim3(2), 256, 0, stream>>>(gsum, 512);
  k_gemm_xw<bf16_t><<<dim3(256, 4), 256, 0, stream>>>(act, W2, b2, xw);
  k_gemm_an<<<dim3(BB * NN, 2), 256, 0, stream>>>(An, xw, pre, gsum, gsumsq);
  k_bnfin<<<dim3(1), 256, 0, stream>>>(gsum, gsumsq, g2, be2, scale, shift);
  k_bnapply<<<dim3(2048), 256, 0, stream>>>(pre, scale, shift, act, nelem);

  // MLP head: graph = act viewed as [64, 131072]
  k_zero<<<dim3(32), 256, 0, stream>>>(z1pre, BB * 128);
  k_head<<<dim3(64), 256, 0, stream>>>(act, Wm1, z1pre);
  k_tail<<<dim3(1), 256, 0, stream>>>(z1pre, bm1, gm1, bem1, Wm2, bm2, gm2, bem2, Wm3, bm3, out);

  (void)in_sizes; (void)n_in; (void)out_size; (void)ws_size;
}
